// EGNNLayer_34574486733397
// MI455X (gfx1250) — matrix operand order repaired and checked by reading, not compiled, not run on hardware
//
#include <hip/hip_runtime.h>
#include <hip/hip_bf16.h>
#include <stdint.h>

// ---------------------------------------------------------------------------
// EGNN layer for MI455X (gfx1250, wave32, WMMA 16x16x32 f16 with f32 accum).
// - h and weights converted to f16 once (halves gather traffic, enables
//   v_wmma_f32_16x16x32_f16); accumulation/scatter/LayerNorm stay f32.
// - Edge kernel: 32 edges/block, 8 waves; each wave owns 16 output columns
//   and two 16-edge M-tiles sharing one B fragment per k-step (2x WMMA/B).
// - Gather staging uses GLOBAL_LOAD_ASYNC_TO_LDS_B128 (ASYNCcnt) via inline
//   asm -- the CDNA5 async memory->LDS path.
// ---------------------------------------------------------------------------

typedef __attribute__((ext_vector_type(16))) _Float16 v16h;
typedef __attribute__((ext_vector_type(8)))  float    v8f;

#define NODE_DIM 128
#define HID      128
#define KPAD1    288   // 2*128 + 3 + 1 = 260 -> pad to 9*32
#define KPADN    256   // node MLP input 128+128
#define EB       32    // edges per block
// LDS strides (u32). Chosen for 16B-aligned rows (async b128 dest) and no
// bank conflicts: 148%64=20 -> 20*r mod 64 distinct for r<16; 132%64=4 ->
// 4*r mod 64 distinct for r<16; msg tiles use 65-u32 stride (odd -> clean).
#define A_STRIDE_U32   148   // edge A tile row: 296 f16 (288 used)
#define M_STRIDE_F16   130   // msg tiles row: 130 f16 (128 used), 65 u32
#define NA_STRIDE_U32  132   // node A tile row: 264 f16 (256 used)

__device__ __forceinline__ float silu(float v) {
    return v / (1.0f + __expf(-v));
}

__device__ __forceinline__ uint32_t pkh(float a, float b) {
    union { _Float16 h[2]; uint32_t u; } r;
    r.h[0] = (_Float16)a; r.h[1] = (_Float16)b;
    return r.u;
}

// CDNA5 async memory->LDS copy, 16 bytes per lane. LDS dest offset is the
// low 32 bits of the flat shared pointer (LDS aperture: addr[31:0] = offset).
__device__ __forceinline__ void async_copy16(void* lds_dst, const void* gsrc) {
    uint32_t l = (uint32_t)(uintptr_t)lds_dst;
    uint64_t g = (uint64_t)(uintptr_t)gsrc;
    asm volatile("global_load_async_to_lds_b128 %0, %1, off"
                 :: "v"(l), "v"(g) : "memory");
}
__device__ __forceinline__ void async_wait0() {
    asm volatile("s_wait_asynccnt 0x0" ::: "memory");
}

// A fragment (16x32 f16, MxK) from LDS, in this file's contiguous k order (same as load_b_frag):
// lane L: M = L&15, half = L>>4; VGPR d -> K = 2d + 16*half.
// (u32 index = K/2)
__device__ __forceinline__ v16h load_a_frag(const uint32_t* sA, int strideU32,
                                            int kstep, int lane) {
    const int m  = lane & 15;
    const int hh = lane >> 4;
    const uint32_t* row = sA + m * strideU32 + kstep * 16;
    union { v16h v; uint32_t u[8]; } r;
#pragma unroll
    for (int d = 0; d < 8; ++d) {
        const int c = d + 8 * hh;  // K/2 within block
        r.u[d] = row[c];
    }
    return r.v;
}

// B fragment (32x16 f16, KxN) from a column-major padded weight matrix
// Wt[col][Kpad]: lane L: N = L&15, half = L>>4; VGPR d -> K = 2d + 16*half.
// 8 contiguous dwords per lane -> global_load_b128 pairs.
__device__ __forceinline__ v16h load_b_frag(const uint32_t* Wt, int kpadU32,
                                            int colBase, int kstep, int lane) {
    const int n  = lane & 15;
    const int hh = lane >> 4;
    const uint32_t* p = Wt + (size_t)(colBase + n) * kpadU32 + kstep * 16 + hh * 8;
    union { v16h v; uint32_t u[8]; } r;
#pragma unroll
    for (int d = 0; d < 8; ++d) r.u[d] = p[d];
    return r.v;
}

__device__ __forceinline__ v8f wmma_f16(v16h a, v16h b, v8f c) {
    return __builtin_amdgcn_wmma_f32_16x16x32_f16(false, a, false, b,
                                                  (short)0, c, false, false);
}

// ---------------------------------------------------------------------------
// Prep kernels
// ---------------------------------------------------------------------------

// dst[col][k] = (k < K) ? src[k][col] : 0, dst is f16 [Ncol][Kpad]
__global__ void transpose_pad_f16_k(const float* __restrict__ src,
                                    _Float16* __restrict__ dst,
                                    int K, int Kpad, int Ncol) {
    int idx = blockIdx.x * blockDim.x + threadIdx.x;
    int total = Ncol * Kpad;
    if (idx >= total) return;
    int col = idx / Kpad, k = idx - col * Kpad;
    dst[(size_t)col * Kpad + k] = (k < K) ? (_Float16)src[(size_t)k * Ncol + col]
                                          : (_Float16)0.0f;
}

__global__ void prep_h_zero_k(const float* __restrict__ h,
                              _Float16* __restrict__ h16,
                              float* __restrict__ agg_msg,
                              float* __restrict__ agg_coord,
                              float* __restrict__ cnt, int N) {
    int total = N * NODE_DIM;
    for (int i = blockIdx.x * blockDim.x + threadIdx.x; i < total;
         i += gridDim.x * blockDim.x) {
        h16[i] = (_Float16)h[i];
        agg_msg[i] = 0.0f;
        if (i < N * 3) agg_coord[i] = 0.0f;
        if (i < N)     cnt[i] = 0.0f;
    }
}

// ---------------------------------------------------------------------------
// Edge kernel: 32 edges/block, 256 threads = 8 waves; wave w owns columns
// [16w,16w+16) over two 16-edge M-tiles.  GEMM1 (288, 9 ksteps) -> SiLU ->
// GEMM2 (128, 4) -> SiLU -> scatter msg; GEMM3 (Wc1) -> SiLU -> dot(Wc2) ->
// coord scatter.  Each k-step loads B once, issues two WMMAs.
// ---------------------------------------------------------------------------
__global__ __launch_bounds__(256) void edge_kernel(
    const _Float16* __restrict__ h16, const float* __restrict__ x,
    const int* __restrict__ ei, const float* __restrict__ eattr,
    const _Float16* __restrict__ W1t, const float* __restrict__ b1,
    const _Float16* __restrict__ W2t, const float* __restrict__ b2,
    const _Float16* __restrict__ Wc1t, const float* __restrict__ bc1,
    const float* __restrict__ Wc2, const float* __restrict__ bc2,
    float* __restrict__ agg_msg, float* __restrict__ agg_coord,
    float* __restrict__ cnt, int E)
{
    __shared__ __align__(16) uint32_t sA[EB * A_STRIDE_U32];
    __shared__ __align__(16) _Float16 sM1[EB * M_STRIDE_F16];
    __shared__ __align__(16) _Float16 sM2[EB * M_STRIDE_F16];
    __shared__ float sCw[EB];
    __shared__ int   sDst[EB];
    __shared__ float sRel[EB][3];

    const int tid  = threadIdx.x;
    const int lane = tid & 31;
    const int wave = tid >> 5;
    const int hh   = lane >> 4;
    const int e0   = blockIdx.x * EB;

    // ---- stage A tiles: [h[src] | h[dst] | edge_attr | dist_sq | 0-pad] ----
#pragma unroll
    for (int p = 0; p < 2; ++p) {
        const int el = p * 16 + (tid >> 4), sub = tid & 15;
        int e = e0 + el; if (e >= E) e = E - 1;          // clamp (dup work ok)
        const int src = ei[e];
        const int dst = ei[E + e];
        const uint32_t* hs = (const uint32_t*)(h16 + (size_t)src * NODE_DIM);
        const uint32_t* hd = (const uint32_t*)(h16 + (size_t)dst * NODE_DIM);
        uint32_t* rowA = sA + el * A_STRIDE_U32;
        // async memory->LDS, 16B per lane per call (ASYNCcnt)
        async_copy16(rowA + sub * 4,      hs + sub * 4);
        async_copy16(rowA + 64 + sub * 4, hd + sub * 4);
        if (sub == 0) {
            sDst[el] = dst;
            const float rx = x[src * 3 + 0] - x[dst * 3 + 0];
            const float ry = x[src * 3 + 1] - x[dst * 3 + 1];
            const float rz = x[src * 3 + 2] - x[dst * 3 + 2];
            sRel[el][0] = rx; sRel[el][1] = ry; sRel[el][2] = rz;
            const float d2 = rx * rx + ry * ry + rz * rz;
            rowA[128] = pkh(eattr[(size_t)e * 3 + 0], eattr[(size_t)e * 3 + 1]);
            rowA[129] = pkh(eattr[(size_t)e * 3 + 2], d2);
#pragma unroll
            for (int c = 130; c < A_STRIDE_U32; ++c) rowA[c] = 0u;
            sCw[el] = 0.0f;
        }
    }
    async_wait0();
    __syncthreads();

    const int colBase = wave * 16;
    const int n = colBase + (lane & 15);
    const uint32_t* sA1 = sA + 16 * A_STRIDE_U32;

    // ---- GEMM1: msg1 = silu(msg_in @ W1 + b1) ------------------------------
    v8f acc0 = {}, acc1 = {};
#pragma unroll
    for (int ks = 0; ks < 9; ++ks) {
        v16h b  = load_b_frag((const uint32_t*)W1t, KPAD1 / 2, colBase, ks, lane);
        v16h a0 = load_a_frag(sA,  A_STRIDE_U32, ks, lane);
        v16h a1 = load_a_frag(sA1, A_STRIDE_U32, ks, lane);
        acc0 = wmma_f16(a0, b, acc0);
        acc1 = wmma_f16(a1, b, acc1);
    }
    {
        const float bias = b1[n];
#pragma unroll
        for (int i = 0; i < 8; ++i) {
            const int m = i + 8 * hh;
            sM1[m * M_STRIDE_F16 + n]        = (_Float16)silu(acc0[i] + bias);
            sM1[(16 + m) * M_STRIDE_F16 + n] = (_Float16)silu(acc1[i] + bias);
        }
    }
    __syncthreads();

    // ---- GEMM2: msg = silu(msg1 @ W2 + b2); scatter agg_msg ----------------
    acc0 = (v8f){}; acc1 = (v8f){};
#pragma unroll
    for (int ks = 0; ks < 4; ++ks) {
        v16h b  = load_b_frag((const uint32_t*)W2t, HID / 2, colBase, ks, lane);
        v16h a0 = load_a_frag((const uint32_t*)sM1, M_STRIDE_F16 / 2, ks, lane);
        v16h a1 = load_a_frag((const uint32_t*)sM1 + 16 * (M_STRIDE_F16 / 2),
                              M_STRIDE_F16 / 2, ks, lane);
        acc0 = wmma_f16(a0, b, acc0);
        acc1 = wmma_f16(a1, b, acc1);
    }
    {
        const float bias = b2[n];
#pragma unroll
        for (int i = 0; i < 8; ++i) {
            const int m = i + 8 * hh;
            const float v0 = silu(acc0[i] + bias);
            const float v1 = silu(acc1[i] + bias);
            sM2[m * M_STRIDE_F16 + n]        = (_Float16)v0;
            sM2[(16 + m) * M_STRIDE_F16 + n] = (_Float16)v1;
            if (e0 + m < E)
                atomicAdd(&agg_msg[(size_t)sDst[m] * HID + n], v0);
            if (e0 + 16 + m < E)
                atomicAdd(&agg_msg[(size_t)sDst[16 + m] * HID + n], v1);
        }
    }
    __syncthreads();

    // ---- GEMM3: c1 = silu(msg @ Wc1 + bc1); coord_w = c1 @ Wc2 + bc2 -------
    acc0 = (v8f){}; acc1 = (v8f){};
#pragma unroll
    for (int ks = 0; ks < 4; ++ks) {
        v16h b  = load_b_frag((const uint32_t*)Wc1t, HID / 2, colBase, ks, lane);
        v16h a0 = load_a_frag((const uint32_t*)sM2, M_STRIDE_F16 / 2, ks, lane);
        v16h a1 = load_a_frag((const uint32_t*)sM2 + 16 * (M_STRIDE_F16 / 2),
                              M_STRIDE_F16 / 2, ks, lane);
        acc0 = wmma_f16(a0, b, acc0);
        acc1 = wmma_f16(a1, b, acc1);
    }
    {
        const float bias = bc1[n];
        const float w2c  = Wc2[n];
#pragma unroll
        for (int i = 0; i < 8; ++i) {
            const int m = i + 8 * hh;
            atomicAdd(&sCw[m],      silu(acc0[i] + bias) * w2c);
            atomicAdd(&sCw[16 + m], silu(acc1[i] + bias) * w2c);
        }
    }
    __syncthreads();

    if (tid < EB && e0 + tid < E) {
        const int m = tid;
        const int dst = sDst[m];
        const float cw = sCw[m] + bc2[0];
        atomicAdd(&agg_coord[(size_t)dst * 3 + 0], sRel[m][0] * cw);
        atomicAdd(&agg_coord[(size_t)dst * 3 + 1], sRel[m][1] * cw);
        atomicAdd(&agg_coord[(size_t)dst * 3 + 2], sRel[m][2] * cw);
        atomicAdd(&cnt[dst], 1.0f);
    }
}

// ---------------------------------------------------------------------------
// Node kernel: 16 nodes per block.  node_in = [h | agg_msg] (256),
// t = silu(node_in @ Wn1 + bn1), h_upd = t @ Wn2 + bn2,
// h_out = LayerNorm(h + h_upd), x_out = x + agg_coord / max(count,1).
// ---------------------------------------------------------------------------
__global__ __launch_bounds__(256) void node_kernel(
    const float* __restrict__ h, const _Float16* __restrict__ h16,
    const float* __restrict__ x,
    const float* __restrict__ agg_msg, const float* __restrict__ agg_coord,
    const float* __restrict__ cnt,
    const _Float16* __restrict__ Wn1t, const float* __restrict__ bn1,
    const _Float16* __restrict__ Wn2t, const float* __restrict__ bn2,
    const float* __restrict__ gamma, const float* __restrict__ beta,
    float* __restrict__ h_out, float* __restrict__ x_out, int N)
{
    __shared__ __align__(16) uint32_t sA[16 * NA_STRIDE_U32];
    __shared__ __align__(16) _Float16 sT[16 * M_STRIDE_F16];
    __shared__ float sY[16][NODE_DIM + 4];
    __shared__ float sMu[16], sRs[16];

    const int tid  = threadIdx.x;
    const int lane = tid & 31;
    const int wave = tid >> 5;
    const int hh   = lane >> 4;
    const int n0   = blockIdx.x * 16;

    // ---- stage A tile: [h16 | f16(agg_msg)] --------------------------------
    {
        const int nl = tid >> 4, sub = tid & 15;
        int node = n0 + nl; if (node >= N) node = N - 1;
        uint32_t* rowA = sA + nl * NA_STRIDE_U32;
        const uint32_t* hp = (const uint32_t*)(h16 + (size_t)node * NODE_DIM);
        async_copy16(rowA + sub * 4, hp + sub * 4);   // async memory->LDS
        const float* am = agg_msg + (size_t)node * HID + sub * 8;
#pragma unroll
        for (int j = 0; j < 4; ++j)
            rowA[64 + sub * 4 + j] = pkh(am[2 * j], am[2 * j + 1]);
    }
    async_wait0();
    __syncthreads();

    const int colBase = wave * 16;
    const int col = colBase + (lane & 15);

    // ---- GEMM Wn1 (K = 256, 8 ksteps) --------------------------------------
    v8f acc = {};
#pragma unroll
    for (int ks = 0; ks < 8; ++ks) {
        v16h a = load_a_frag(sA, NA_STRIDE_U32, ks, lane);
        v16h b = load_b_frag((const uint32_t*)Wn1t, KPADN / 2, colBase, ks, lane);
        acc = wmma_f16(a, b, acc);
    }
    {
        const float bias = bn1[col];
#pragma unroll
        for (int i = 0; i < 8; ++i) {
            const int m = i + 8 * hh;
            sT[m * M_STRIDE_F16 + col] = (_Float16)silu(acc[i] + bias);
        }
    }
    __syncthreads();

    // ---- GEMM Wn2 (K = 128, 4 ksteps); y = h + h_upd -----------------------
    acc = (v8f){};
#pragma unroll
    for (int ks = 0; ks < 4; ++ks) {
        v16h a = load_a_frag((const uint32_t*)sT, M_STRIDE_F16 / 2, ks, lane);
        v16h b = load_b_frag((const uint32_t*)Wn2t, HID / 2, colBase, ks, lane);
        acc = wmma_f16(a, b, acc);
    }
    {
        const float bias = bn2[col];
#pragma unroll
        for (int i = 0; i < 8; ++i) {
            const int m = i + 8 * hh;
            int node = n0 + m; if (node >= N) node = N - 1;
            sY[m][col] = h[(size_t)node * NODE_DIM + col] + acc[i] + bias;
        }
    }
    __syncthreads();

    // ---- LayerNorm stats + x_out (one thread per node row) -----------------
    if (tid < 16) {
        const int m = tid;
        float s = 0.0f;
        for (int c = 0; c < NODE_DIM; ++c) s += sY[m][c];
        const float mu = s * (1.0f / NODE_DIM);
        float v = 0.0f;
        for (int c = 0; c < NODE_DIM; ++c) {
            const float d = sY[m][c] - mu;
            v += d * d;
        }
        sMu[m] = mu;
        sRs[m] = rsqrtf(v * (1.0f / NODE_DIM) + 1e-5f);

        const int node = n0 + m;
        if (node < N) {
            const float c = fmaxf(cnt[node], 1.0f);
#pragma unroll
            for (int j = 0; j < 3; ++j)
                x_out[(size_t)node * 3 + j] =
                    x[(size_t)node * 3 + j] + agg_coord[(size_t)node * 3 + j] / c;
        }
    }
    __syncthreads();

    // ---- write h_out -------------------------------------------------------
    {
        const int m = tid >> 4;
        const int node = n0 + m;
        if (node < N) {
            const int c0 = (tid & 15) * 8;
            const float mu = sMu[m], rs = sRs[m];
#pragma unroll
            for (int j = 0; j < 8; ++j) {
                const int c = c0 + j;
                h_out[(size_t)node * NODE_DIM + c] =
                    (sY[m][c] - mu) * rs * gamma[c] + beta[c];
            }
        }
    }
}

// ---------------------------------------------------------------------------
// Host launcher
// ---------------------------------------------------------------------------
static inline size_t alignup(size_t v, size_t a) { return (v + a - 1) & ~(a - 1); }

extern "C" void kernel_launch(void* const* d_in, const int* in_sizes, int n_in,
                              void* d_out, int out_size, void* d_ws, size_t ws_size,
                              hipStream_t stream) {
    const float* h     = (const float*)d_in[0];
    const float* x     = (const float*)d_in[1];
    const int*   ei    = (const int*)  d_in[2];
    const float* eattr = (const float*)d_in[3];
    const float* W1  = (const float*)d_in[4];  const float* b1  = (const float*)d_in[5];
    const float* W2  = (const float*)d_in[6];  const float* b2  = (const float*)d_in[7];
    const float* Wc1 = (const float*)d_in[8];  const float* bc1 = (const float*)d_in[9];
    const float* Wc2 = (const float*)d_in[10]; const float* bc2 = (const float*)d_in[11];
    const float* Wn1 = (const float*)d_in[12]; const float* bn1 = (const float*)d_in[13];
    const float* Wn2 = (const float*)d_in[14]; const float* bn2 = (const float*)d_in[15];
    const float* gamma = (const float*)d_in[16];
    const float* beta  = (const float*)d_in[17];

    const int N = in_sizes[0] / NODE_DIM;
    const int E = in_sizes[2] / 2;

    // workspace carve-up
    char* ws = (char*)d_ws;
    size_t off = 0;
    auto take = [&](size_t bytes) { char* p = ws + off; off = alignup(off + bytes, 256); return p; };
    _Float16* h16  = (_Float16*)take((size_t)N * NODE_DIM * 2);
    _Float16* W1t  = (_Float16*)take((size_t)HID * KPAD1 * 2);
    _Float16* W2t  = (_Float16*)take((size_t)HID * HID * 2);
    _Float16* Wc1t = (_Float16*)take((size_t)HID * HID * 2);
    _Float16* Wn1t = (_Float16*)take((size_t)HID * KPADN * 2);
    _Float16* Wn2t = (_Float16*)take((size_t)NODE_DIM * HID * 2);
    float* agg_msg   = (float*)take((size_t)N * HID * 4);
    float* agg_coord = (float*)take((size_t)N * 3 * 4);
    float* cnt       = (float*)take((size_t)N * 4);

    float* h_out = (float*)d_out;
    float* x_out = h_out + (size_t)N * NODE_DIM;

    // --- prep: weight transposes (f16, column-major, K padded) --------------
    {
        int t;
        t = HID * KPAD1;
        transpose_pad_f16_k<<<(t + 255) / 256, 256, 0, stream>>>(W1, W1t, 2 * NODE_DIM + 4, KPAD1, HID);
        t = HID * HID;
        transpose_pad_f16_k<<<(t + 255) / 256, 256, 0, stream>>>(W2, W2t, HID, HID, HID);
        transpose_pad_f16_k<<<(t + 255) / 256, 256, 0, stream>>>(Wc1, Wc1t, HID, HID, HID);
        t = HID * KPADN;
        transpose_pad_f16_k<<<(t + 255) / 256, 256, 0, stream>>>(Wn1, Wn1t, KPADN, KPADN, HID);
        t = NODE_DIM * HID;
        transpose_pad_f16_k<<<(t + 255) / 256, 256, 0, stream>>>(Wn2, Wn2t, HID, HID, NODE_DIM);
    }
    // --- prep: h -> f16, zero accumulators ----------------------------------
    {
        int total = N * NODE_DIM;
        int blocks = (total + 255) / 256;
        prep_h_zero_k<<<blocks, 256, 0, stream>>>(h, h16, agg_msg, agg_coord, cnt, N);
    }
    // --- edge phase ---------------------------------------------------------
    edge_kernel<<<(E + EB - 1) / EB, 256, 0, stream>>>(
        h16, x, ei, eattr, W1t, b1, W2t, b2, Wc1t, bc1, Wc2, bc2,
        agg_msg, agg_coord, cnt, E);
    // --- node phase ---------------------------------------------------------
    node_kernel<<<(N + 15) / 16, 256, 0, stream>>>(
        h, h16, x, agg_msg, agg_coord, cnt, Wn1t, bn1, Wn2t, bn2,
        gamma, beta, h_out, x_out, N);
}
